// SelfAttn_31396210933925
// MI455X (gfx1250) — matrix operand order repaired and checked by reading, not compiled, not run on hardware
//
#include <hip/hip_runtime.h>

// ---------------------------------------------------------------------------
// Linear self-attention for MI455X (gfx1250, wave32, WMMA).
// Precision: bf16 WMMA (16x16x32) with f32 accumulate for all matrix ops.
// Round 3: B tile of the big GEMM staged through LDS cooperatively
// (8 waves/block share one 64-col W group), double-buffered K-chunks of 128
// using GLOBAL_LOAD_ASYNC_TO_LDS_B128 (ASYNCcnt) + s_wait_asynccnt,
// padded LDS layout (row stride 68 dwords) to avoid bank conflicts.
// ---------------------------------------------------------------------------

typedef __attribute__((ext_vector_type(16))) __bf16 v16bf;
typedef __attribute__((ext_vector_type(8)))  __bf16 v8bf;
typedef __attribute__((ext_vector_type(8)))  float  v8f;

#define BATCH   2
#define SEQ     8192
#define DIMM    1024
#define NH      16
#define HD      64
#define MROWS   (BATCH * SEQ)     // 16384
#define KDIM    DIMM              // 1024
#define GROUPS  48                // 3*DIMM / 64 column groups
#define CHUNK   1024              // n-chunk for kv reduction
#define KC      128               // K-chunk staged in LDS
#define BPAD    136               // padded LDS row stride (halves): 68 dwords

union Frag16 { v16bf v; v8bf h[2]; __bf16 e[16]; };

// Async global->LDS copy, 16 bytes per lane.  vdst = LDS byte offset VGPR,
// vaddr = 64-bit global address.  Tracked by ASYNCcnt.
static __device__ __forceinline__ void async_b128(unsigned ldsoff,
                                                  const __bf16* gp) {
    asm volatile("global_load_async_to_lds_b128 %0, %1, off"
                 :: "v"(ldsoff), "v"((unsigned long long)(uintptr_t)gp)
                 : "memory");
}
static __device__ __forceinline__ void wait_async0() {
    asm volatile("s_wait_asynccnt 0" ::: "memory");
}
static __device__ __forceinline__ unsigned lds_off(const void* p) {
    return (unsigned)(uintptr_t)p;   // ISA: LDS generic addr low 32b = offset
}

// ---------------------------------------------------------------------------
// fp32 -> bf16 convert (grid-stride)
// ---------------------------------------------------------------------------
__global__ void cvt_f32_bf16(const float* __restrict__ in,
                             __bf16* __restrict__ out, int n) {
    int i = blockIdx.x * blockDim.x + threadIdx.x;
    int stride = gridDim.x * blockDim.x;
    for (; i < n; i += stride) out[i] = (__bf16)in[i];
}

// ---------------------------------------------------------------------------
// Stage 1: qkv = x @ W^T with fused per-head l2norm + softplus epilogue.
// Block = 8 waves sharing one 64-column group g; wave w takes row tile
// mt = blockRow*8 + w (16 rows).  Wave computes 16(M) x 64(N).
// B (64 W rows x KC halves) staged in LDS, double buffered, async copies.
//   g <  16 : q head g    -> qn [bh, n, d]   (row-major d, stage-3 A)
//   g < 32  : k head g-16 -> kT [bh, d, n]   (transposed, stage-2 A)
//   else    : v head g-32 -> vT [bh, d, n]   (transposed, stage-2 B)
// A-frag: lane holds row m=lane%16, K-halves base (lane>>4)*8 and +16.
// B-frag: lane holds col n=lane%16 (a W row), K-halves base (lane>>4)*8 and +16, as A.
// ---------------------------------------------------------------------------
__global__ __launch_bounds__(256) void qkv_gemm_act(
    const __bf16* __restrict__ xb, const __bf16* __restrict__ wb,
    __bf16* __restrict__ qn, __bf16* __restrict__ kT, __bf16* __restrict__ vT)
{
    __shared__ __bf16 Bs[2][64 * BPAD];          // 2 x 17 KB

    const int tid  = threadIdx.x;
    const int lane = tid & 31;
    const int w    = tid >> 5;                   // 0..7
    const int g    = blockIdx.x % GROUPS;        // shared by whole block
    const int mt   = (blockIdx.x / GROUPS) * 8 + w;

    const int l16  = lane & 15;
    const int half = lane >> 4;
    const int m0   = mt * 16;

    const __bf16* arow = xb + (size_t)(m0 + l16) * KDIM + half * 8;

    // Cooperative stage of one KC-chunk of B: thread t copies 64 B
    // (row r = t>>2, halves [part*32, part*32+32) of the chunk).
    const int srow  = tid >> 2;                  // 0..63
    const int spart = tid & 3;                   // 0..3
    const __bf16* ssrc0 = wb + (size_t)(g * 64 + srow) * KDIM + spart * 32;

    v8f acc[4] = {};

    // prime buffer 0 with chunk kc=0
    {
        const __bf16* s = ssrc0;
        unsigned d = lds_off(&Bs[0][srow * BPAD + spart * 32]);
#pragma unroll
        for (int j = 0; j < 4; ++j) async_b128(d + j * 16, s + j * 8);
    }

    int buf = 0;
    for (int kc = 0; kc < KDIM; kc += KC) {
        wait_async0();
        __syncthreads();                         // chunk `buf` ready; prior
                                                 // readers of buf^1 finished
        if (kc + KC < KDIM) {                    // prefetch next chunk
            const __bf16* s = ssrc0 + kc + KC;
            unsigned d = lds_off(&Bs[buf ^ 1][srow * BPAD + spart * 32]);
#pragma unroll
            for (int j = 0; j < 4; ++j) async_b128(d + j * 16, s + j * 8);
        }
#pragma unroll
        for (int ks = 0; ks < KC; ks += 32) {
            Frag16 A;
            A.h[0] = *(const v8bf*)(arow + kc + ks);
            A.h[1] = *(const v8bf*)(arow + kc + ks + 16);
#pragma unroll
            for (int ct = 0; ct < 4; ++ct) {
                Frag16 Bv;
                const __bf16* bp = &Bs[buf][(ct * 16 + l16) * BPAD
                                            + ks + half * 8];
                Bv.h[0] = *(const v8bf*)(bp);
                Bv.h[1] = *(const v8bf*)(bp + 16);
                acc[ct] = __builtin_amdgcn_wmma_f32_16x16x32_bf16(
                    false, A.v, false, Bv.v, (short)0, acc[ct], false, false);
            }
        }
        buf ^= 1;
    }

    const int b     = m0 / SEQ;
    const int nbase = m0 % SEQ;

    if (g < 32) {                 // q or k: l2-normalize over d=64, softplus
        const bool isq = (g < 16);
        const int  h   = isq ? g : g - 16;
        const size_t bh = (size_t)b * NH + h;
#pragma unroll
        for (int r = 0; r < 8; ++r) {
            float s = acc[0][r]*acc[0][r] + acc[1][r]*acc[1][r]
                    + acc[2][r]*acc[2][r] + acc[3][r]*acc[3][r];
            s += __shfl_xor(s, 1);
            s += __shfl_xor(s, 2);
            s += __shfl_xor(s, 4);
            s += __shfl_xor(s, 8);   // row sum-of-squares over all 64 cols
            const float inv = 1.0f / fmaxf(sqrtf(s), 1e-12f);
            const int m = r + 8 * half;
            const int n = nbase + m;
#pragma unroll
            for (int ct = 0; ct < 4; ++ct) {
                float val = acc[ct][r] * inv;
                val = (val > 20.0f) ? val : log1pf(expf(val));  // softplus
                const int col = ct * 16 + l16;
                if (isq) qn[(bh * SEQ + n) * HD + col] = (__bf16)val;
                else     kT[(bh * HD + col) * SEQ + n] = (__bf16)val;
            }
        }
    } else {                      // v: raw copy, transposed layout
        const int h = g - 32;
        const size_t bh = (size_t)b * NH + h;
#pragma unroll
        for (int r = 0; r < 8; ++r) {
            const int m = r + 8 * half;
            const int n = nbase + m;
#pragma unroll
            for (int ct = 0; ct < 4; ++ct) {
                const int col = ct * 16 + l16;
                vT[(bh * HD + col) * SEQ + n] = (__bf16)acc[ct][r];
            }
        }
    }
}

// ---------------------------------------------------------------------------
// ksum[bh][d] = sum_n kT[bh][d][n] + EPS   (one block per bh, LDS reduce)
// ---------------------------------------------------------------------------
__global__ __launch_bounds__(256) void ksum_kernel(
    const __bf16* __restrict__ kT, float* __restrict__ ksum)
{
    __shared__ float sred[256];
    const int bh   = blockIdx.x;
    const int d    = threadIdx.x & 63;
    const int part = threadIdx.x >> 6;          // 0..3
    const __bf16* row = kT + ((size_t)bh * HD + d) * SEQ + part * (SEQ / 4);
    float s = 0.f;
    for (int i = 0; i < SEQ / 4; ++i) s += (float)row[i];
    sred[threadIdx.x] = s;
    __syncthreads();
    if (part == 0)
        ksum[bh * HD + d] = sred[d] + sred[64 + d] + sred[128 + d]
                          + sred[192 + d] + 1e-6f;
}

// ---------------------------------------------------------------------------
// Stage 2: kvT[bh][c][m] += scale * sum_n kT[bh][m][n] * vT[bh][c][n]
// Split-n (CHUNK rows per block) + fp32 atomics. 8 waves cover the 4x4
// fragment grid of the 64x64 output (2 fragments/wave). Double-buffered.
// Stored transposed (c-major) so stage-3 B lanes read contiguously.
// ---------------------------------------------------------------------------
__global__ __launch_bounds__(256) void kv_accum(
    const __bf16* __restrict__ kT, const __bf16* __restrict__ vT,
    float* __restrict__ kvT)
{
    const int lane  = threadIdx.x & 31;
    const int w     = threadIdx.x >> 5;         // 0..7
    const int bh    = blockIdx.x / (SEQ / CHUNK);
    const int chunk = blockIdx.x % (SEQ / CHUNK);
    const int l16 = lane & 15, half = lane >> 4;
    const int mt  = w & 3;
    const int ct0 = (w >> 2) * 2;

    const int n0 = chunk * CHUNK;
    const __bf16* arow  = kT + ((size_t)bh * HD + mt * 16 + l16) * SEQ
                        + n0 + half * 8;
    const __bf16* brow0 = vT + ((size_t)bh * HD + ct0 * 16 + l16) * SEQ
                        + n0 + half * 8;

    v8f acc[2] = {};

    Frag16 Ac, Bc[2];
    Ac.h[0] = *(const v8bf*)(arow);
    Ac.h[1] = *(const v8bf*)(arow + 16);
#pragma unroll
    for (int c = 0; c < 2; ++c) {
        const __bf16* bp = brow0 + (size_t)(c * 16) * SEQ;
        Bc[c].h[0] = *(const v8bf*)bp;
        Bc[c].h[1] = *(const v8bf*)(bp + 16);
    }
    for (int kk = 0; kk < CHUNK; kk += 32) {
        const int kn = (kk + 32) & (CHUNK - 1);        // wrap: always valid
        Frag16 An, Bn[2];
        An.h[0] = *(const v8bf*)(arow + kn);
        An.h[1] = *(const v8bf*)(arow + kn + 16);
#pragma unroll
        for (int c = 0; c < 2; ++c) {
            const __bf16* bp = brow0 + (size_t)(c * 16) * SEQ + kn;
            Bn[c].h[0] = *(const v8bf*)bp;
            Bn[c].h[1] = *(const v8bf*)(bp + 16);
        }
#pragma unroll
        for (int c = 0; c < 2; ++c)
            acc[c] = __builtin_amdgcn_wmma_f32_16x16x32_bf16(
                false, Ac.v, false, Bc[c].v, (short)0, acc[c], false, false);
        Ac = An;
#pragma unroll
        for (int c = 0; c < 2; ++c) Bc[c] = Bn[c];
    }

    const float scale = 0.125f;                 // 64^-0.5
#pragma unroll
    for (int c = 0; c < 2; ++c)
#pragma unroll
        for (int r = 0; r < 8; ++r) {
            const int m  = mt * 16 + r + 8 * half;
            const int cc = (ct0 + c) * 16 + l16;
            atomicAdd(&kvT[((size_t)bh * HD + cc) * HD + m], acc[c][r] * scale);
        }
}

// ---------------------------------------------------------------------------
// Stage 3: out = (q @ kv) / (q . ksum). One wave = 16(n) x 64(d) tile.
// Denominators computed from the in-register A fragments + shfl.
// ---------------------------------------------------------------------------
__global__ __launch_bounds__(256) void attn_out(
    const __bf16* __restrict__ qn, const float* __restrict__ kvT,
    const float* __restrict__ ksum, float* __restrict__ out)
{
    const int lane = threadIdx.x & 31;
    const int wave = blockIdx.x * 8 + (threadIdx.x >> 5);
    const int bh = wave / (SEQ / 16);
    const int nt = wave % (SEQ / 16);
    const int l16 = lane & 15, half = lane >> 4;
    const int b = bh / NH, h = bh % NH;

    const __bf16* arow  = qn + ((size_t)bh * SEQ + nt * 16 + l16) * HD + half * 8;
    const float*  ksrow = ksum + bh * HD;

    Frag16 A[2];
    float partial = 0.f;
#pragma unroll
    for (int ks = 0; ks < 2; ++ks) {
        A[ks].h[0] = *(const v8bf*)(arow + ks * 32);
        A[ks].h[1] = *(const v8bf*)(arow + ks * 32 + 16);
#pragma unroll
        for (int e = 0; e < 16; ++e) {          // A-layout d index for elem e
            const int p = e >> 1;
            const int d = ks * 32 + half * 8 + ((p < 4) ? 0 : 16)
                        + (p & 3) * 2 + (e & 1);
            partial += (float)A[ks].e[e] * ksrow[d];
        }
    }
    partial += __shfl_xor(partial, 16);         // lane l (l<16): denom of row l

    v8f acc[4] = {};
#pragma unroll
    for (int ks = 0; ks < 2; ++ks) {
#pragma unroll
        for (int ct = 0; ct < 4; ++ct) {
            Frag16 Bv;                          // B elem e: k = ks*32+half*8+e (+8 if e>=8)
            const float* kvrow = kvT + ((size_t)bh * HD + ct * 16 + l16) * HD
                               + ks * 32 + half * 8;
#pragma unroll
            for (int e = 0; e < 16; ++e) Bv.e[e] = (__bf16)kvrow[e < 8 ? e : e + 8];
            acc[ct] = __builtin_amdgcn_wmma_f32_16x16x32_bf16(
                false, A[ks].v, false, Bv.v, (short)0, acc[ct], false, false);
        }
    }
#pragma unroll
    for (int r = 0; r < 8; ++r) {
        const int m = r + 8 * half;
        const float nrm = 1.0f / __shfl(partial, m);
        const int n = nt * 16 + m;
#pragma unroll
        for (int ct = 0; ct < 4; ++ct) {
            const int col = h * HD + ct * 16 + l16;
            out[((size_t)b * SEQ + n) * DIMM + col] = acc[ct][r] * nrm;
        }
    }
}

// ---------------------------------------------------------------------------
extern "C" void kernel_launch(void* const* d_in, const int* in_sizes, int n_in,
                              void* d_out, int out_size, void* d_ws, size_t ws_size,
                              hipStream_t stream) {
    (void)in_sizes; (void)n_in; (void)out_size; (void)ws_size;
    const float* x = (const float*)d_in[0];
    const float* W = (const float*)d_in[1];
    float* out = (float*)d_out;

    char* ws = (char*)d_ws;
    size_t off = 0;
    auto walloc = [&](size_t bytes) -> void* {
        void* p = ws + off;
        off += bytes;
        off = (off + 255) & ~(size_t)255;
        return p;
    };
    __bf16* xb  = (__bf16*)walloc((size_t)MROWS * KDIM * 2);       // 33.5 MB
    __bf16* wb  = (__bf16*)walloc((size_t)3 * DIMM * KDIM * 2);    //  6.3 MB
    __bf16* qn  = (__bf16*)walloc((size_t)BATCH * NH * SEQ * HD * 2);
    __bf16* kT  = (__bf16*)walloc((size_t)BATCH * NH * SEQ * HD * 2);
    __bf16* vT  = (__bf16*)walloc((size_t)BATCH * NH * SEQ * HD * 2);
    float*  kvT = (float*)walloc((size_t)BATCH * NH * HD * HD * 4);
    float*  ksm = (float*)walloc((size_t)BATCH * NH * HD * 4);

    hipMemsetAsync(kvT, 0, (size_t)BATCH * NH * HD * HD * 4, stream);

    cvt_f32_bf16<<<2048, 256, 0, stream>>>(x, xb, MROWS * KDIM);
    cvt_f32_bf16<<<1024, 256, 0, stream>>>(W, wb, 3 * DIMM * KDIM);

    qkv_gemm_act<<<(MROWS / 16 / 8) * GROUPS, 256, 0, stream>>>(xb, wb, qn, kT, vT);
    ksum_kernel<<<BATCH * NH, 256, 0, stream>>>(kT, ksm);
    kv_accum<<<BATCH * NH * (SEQ / CHUNK), 256, 0, stream>>>(kT, vT, kvT);
    attn_out<<<BATCH * NH * (SEQ / 16) / 8, 256, 0, stream>>>(qn, kvT, ksm, out);
}
